// FusedRoutedMLP_35175782154749
// MI455X (gfx1250) — matrix operand order repaired and checked by reading, not compiled, not run on hardware
//
#include <hip/hip_runtime.h>
#include <cstdint>
#include <cstddef>

// Problem constants (match reference)
static constexpr int kE = 8;
static constexpr int kH = 1024;    // hidden
static constexpr int kI = 1408;    // intermediate
static constexpr int kN1 = 2 * kI; // 2816, w1 row width
static constexpr int kT = 2048;
static constexpr int kK = 2;
static constexpr int MAXT = 1024;  // cap on tokens per expert (mean=512)

static constexpr int NT1  = kN1 / 16;  // 176 n-tiles in w1
static constexpr int NT2  = kH / 16;   // 64  n-tiles in w2
static constexpr int KBC1 = kH / 32;   // 32  k-blocks in w1
static constexpr int KBC2 = kI / 32;   // 44  k-blocks in w2

typedef __attribute__((ext_vector_type(16))) __bf16 bf16x16;
typedef __attribute__((ext_vector_type(8)))  __bf16 bf16x8;
typedef __attribute__((ext_vector_type(8)))  float  f32x8;

__device__ __forceinline__ __bf16 f2bf(float f) {
  uint32_t u = __builtin_bit_cast(uint32_t, f);
  u += 0x7FFFu + ((u >> 16) & 1u);          // round-to-nearest-even
  return __builtin_bit_cast(__bf16, (uint16_t)(u >> 16));
}

__device__ __forceinline__ uint32_t pk2(float a, float b) {
  return (uint32_t)__builtin_bit_cast(uint16_t, f2bf(a)) |
         ((uint32_t)__builtin_bit_cast(uint16_t, f2bf(b)) << 16);
}

__device__ __forceinline__ bf16x16 cat16(bf16x8 lo, bf16x8 hi) {
  return __builtin_shufflevector(lo, hi, 0,1,2,3,4,5,6,7,8,9,10,11,12,13,14,15);
}

// ---------------------------------------------------------------------------
// 1) Deterministic routing: one thread per expert scans all (t,k) pairs.
// ---------------------------------------------------------------------------
__global__ void route_kernel(const long long* __restrict__ ids,
                             const float* __restrict__ rw,
                             int* __restrict__ counts,
                             int* __restrict__ ilist,   // i = t*K + k
                             float* __restrict__ wlist) {
  int e = threadIdx.x;
  if (e >= kE) return;
  int cnt = 0;
  for (int i = 0; i < kT * kK; ++i) {
    if ((int)ids[i] == e) {
      if (cnt < MAXT) {
        ilist[e * MAXT + cnt] = i;
        wlist[e * MAXT + cnt] = rw[i];
        ++cnt;
      }
    }
  }
  counts[e] = cnt;
}

// ---------------------------------------------------------------------------
// 2a) Pack x -> bf16 row-major.
// ---------------------------------------------------------------------------
__global__ __launch_bounds__(256)
void pack_x_kernel(const float* __restrict__ x, __bf16* __restrict__ px) {
  int i = blockIdx.x * 256 + threadIdx.x;        // 4 floats per thread
  if (i >= kT * kH / 4) return;
  float4 v = ((const float4*)x)[i];
  uint2 o = { pk2(v.x, v.y), pk2(v.z, v.w) };
  ((uint2*)px)[i] = o;
}

// ---------------------------------------------------------------------------
// 2b) Pack weights -> bf16 fragment-direct layout.
//     Chunk ((e*nt_count + nt)*kbc + kb)*32 + lane holds the 16 bf16 this
//     lane feeds to v_wmma as B: n = nt*16 + (lane&15),
//     k = kb*32 + (lane>>4)*16 + j  (j = 0..15).
// ---------------------------------------------------------------------------
__global__ __launch_bounds__(256)
void pack_w_kernel(const float* __restrict__ w, __bf16* __restrict__ pw,
                   int ncols, int nt_count, int kbc) {
  int gid = blockIdx.x * 256 + threadIdx.x;
  int total = kE * nt_count * kbc * 32;
  if (gid >= total) return;
  int lane = gid & 31;
  int kb   = (gid >> 5) % kbc;
  int nt   = ((gid >> 5) / kbc) % nt_count;
  int e    = gid / (32 * kbc * nt_count);
  int n    = nt * 16 + (lane & 15);
  int kbase = kb * 32 + (lane >> 4) * 16;
  int rows_per_e = kbc * 32;
  const float* src = w + ((size_t)e * rows_per_e + kbase) * ncols + n;
  bf16x16 v;
#pragma unroll
  for (int j = 0; j < 16; ++j) v[j] = f2bf(src[(size_t)j * ncols]);
  *(bf16x16*)(pw + (size_t)gid * 16) = v;
}

// ---------------------------------------------------------------------------
// 3) GEMM1 + SiLU*up. Block: M=32 tokens, N=128 gate cols + matching up cols.
//    8 waves: wave wn owns gate n-tile wn and up n-tile wn, BOTH m-tiles
//    (B fragment reused for 2 WMMAs). A double-buffered in LDS.
// ---------------------------------------------------------------------------
__global__ __launch_bounds__(256)
void gemm1_silu_kernel(const __bf16* __restrict__ px,
                       const bf16x16* __restrict__ pw1c,
                       const int* __restrict__ counts,
                       const int* __restrict__ ilist,
                       const float* __restrict__ wlist,
                       __bf16* __restrict__ act) {
  constexpr int KB = 64, LSA = 72;
  __shared__ alignas(16) __bf16 As[2][32][LSA];
  __shared__ float Wts[32];

  const int e = blockIdx.z;
  int cnt = counts[e]; if (cnt > MAXT) cnt = MAXT;
  const int m_base = blockIdx.y * 32;
  if (m_base >= cnt) return;
  const int nn = blockIdx.x * 128;               // gate column base in [0, I)

  const int tid  = threadIdx.x;
  const int lane = tid & 31;
  const int wn   = tid >> 5;                     // 0..7: n-tile within block
  const int idx  = lane & 15;
  const int half = lane >> 4;

  if (tid < 32) {
    int slot = m_base + tid;
    Wts[tid] = (slot < cnt) ? wlist[e * MAXT + slot] : 0.f;
  }

  // A-tile loader: row pointer fixed across the k-loop (hoisted to registers).
  const int lr = tid >> 3;                       // loader row 0..31
  const int lc = (tid & 7) << 3;                 // 8 bf16 per thread
  const int lslot = (m_base + lr < cnt) ? (m_base + lr) : 0;
  const int ltok  = ilist[e * MAXT + lslot] / kK;
  const __bf16* xrow = px + (size_t)ltok * kH + lc;

  // Per-wave B fragment bases (chunk units of 16 bf16 = 32B)
  const int ntg = (nn >> 4) + wn;
  const int ntu = ((kI + nn) >> 4) + wn;
  const bf16x16* bgp = pw1c + ((size_t)(e * NT1 + ntg) * KBC1) * 32 + lane;
  const bf16x16* bup = pw1c + ((size_t)(e * NT1 + ntu) * KBC1) * 32 + lane;

  f32x8 accg0 = {}, accg1 = {};
  f32x8 accu0 = {}, accu1 = {};

  uint4 av = *(const uint4*)(xrow);              // prefetch tile 0
  int buf = 0;
  for (int kk = 0; kk < kH; kk += KB) {
    *(uint4*)&As[buf][lr][lc] = av;
    __syncthreads();
    av = *(const uint4*)(xrow + kk + KB);        // prefetch next (overlaps WMMA)
#pragma unroll
    for (int s = 0; s < 2; ++s) {
      int kb = (kk >> 5) + s;
      bf16x8 a0l = *(const bf16x8*)&As[buf][idx][s * 32 + half * 16];
      bf16x8 a0h = *(const bf16x8*)&As[buf][idx][s * 32 + half * 16 + 8];
      bf16x16 a0 = cat16(a0l, a0h);
      bf16x8 a1l = *(const bf16x8*)&As[buf][16 + idx][s * 32 + half * 16];
      bf16x8 a1h = *(const bf16x8*)&As[buf][16 + idx][s * 32 + half * 16 + 8];
      bf16x16 a1 = cat16(a1l, a1h);
      bf16x16 bg = bgp[(size_t)kb * 32];
      bf16x16 bu = bup[(size_t)kb * 32];
      accg0 = __builtin_amdgcn_wmma_f32_16x16x32_bf16(false, a0, false, bg,
                                                      (short)0, accg0, false, false);
      accg1 = __builtin_amdgcn_wmma_f32_16x16x32_bf16(false, a1, false, bg,
                                                      (short)0, accg1, false, false);
      accu0 = __builtin_amdgcn_wmma_f32_16x16x32_bf16(false, a0, false, bu,
                                                      (short)0, accu0, false, false);
      accu1 = __builtin_amdgcn_wmma_f32_16x16x32_bf16(false, a1, false, bu,
                                                      (short)0, accu1, false, false);
    }
    buf ^= 1;
  }
  __syncthreads();

  const int col = nn + wn * 16 + idx;
#pragma unroll
  for (int mt = 0; mt < 2; ++mt) {
    const f32x8& ag = mt ? accg1 : accg0;
    const f32x8& au = mt ? accu1 : accu0;
#pragma unroll
    for (int r = 0; r < 8; ++r) {
      int m = mt * 16 + r + half * 8;
      int slot = m_base + m;
      if (slot < cnt) {
        float g = ag[r];
        float u = au[r];
        float s = g * __builtin_amdgcn_rcpf(1.f + __expf(-g));   // fast SiLU
        float v = s * u * Wts[m];
        act[(size_t)(e * MAXT + slot) * kI + col] = f2bf(v);
      }
    }
  }
}

// ---------------------------------------------------------------------------
// 4) GEMM2: y = act(bf16) x w2. Block: M=32, N=128; wave wn owns one n-tile,
//    both m-tiles (B fragment reused for 2 WMMAs).
// ---------------------------------------------------------------------------
__global__ __launch_bounds__(256)
void gemm2_scatter_kernel(const __bf16* __restrict__ act,
                          const bf16x16* __restrict__ pw2c,
                          const int* __restrict__ counts,
                          const int* __restrict__ ilist,
                          float* __restrict__ yws) {
  constexpr int KB = 64, LSA = 72;
  __shared__ alignas(16) __bf16 As[2][32][LSA];
  __shared__ int Itk[32];

  const int e = blockIdx.z;
  int cnt = counts[e]; if (cnt > MAXT) cnt = MAXT;
  const int m_base = blockIdx.y * 32;
  if (m_base >= cnt) return;
  const int nn = blockIdx.x * 128;

  const int tid  = threadIdx.x;
  const int lane = tid & 31;
  const int wn   = tid >> 5;
  const int idx  = lane & 15;
  const int half = lane >> 4;

  if (tid < 32) {
    int slot = m_base + tid;
    Itk[tid] = (slot < cnt) ? ilist[e * MAXT + slot] : 0;
  }

  const int lr = tid >> 3;
  const int lc = (tid & 7) << 3;
  const int lslot = (m_base + lr < cnt) ? (m_base + lr) : 0;
  const __bf16* arow = act + (size_t)(e * MAXT + lslot) * kI + lc;

  const int nt = (nn >> 4) + wn;
  const bf16x16* bp = pw2c + ((size_t)(e * NT2 + nt) * KBC2) * 32 + lane;

  f32x8 acc0 = {}, acc1 = {};

  uint4 av = *(const uint4*)(arow);
  int buf = 0;
  for (int kk = 0; kk < kI; kk += KB) {
    *(uint4*)&As[buf][lr][lc] = av;
    __syncthreads();
    av = *(const uint4*)(arow + kk + KB);
#pragma unroll
    for (int s = 0; s < 2; ++s) {
      int kb = (kk >> 5) + s;
      bf16x8 a0l = *(const bf16x8*)&As[buf][idx][s * 32 + half * 16];
      bf16x8 a0h = *(const bf16x8*)&As[buf][idx][s * 32 + half * 16 + 8];
      bf16x16 a0 = cat16(a0l, a0h);
      bf16x8 a1l = *(const bf16x8*)&As[buf][16 + idx][s * 32 + half * 16];
      bf16x8 a1h = *(const bf16x8*)&As[buf][16 + idx][s * 32 + half * 16 + 8];
      bf16x16 a1 = cat16(a1l, a1h);
      bf16x16 b = bp[(size_t)kb * 32];
      acc0 = __builtin_amdgcn_wmma_f32_16x16x32_bf16(false, a0, false, b,
                                                     (short)0, acc0, false, false);
      acc1 = __builtin_amdgcn_wmma_f32_16x16x32_bf16(false, a1, false, b,
                                                     (short)0, acc1, false, false);
    }
    buf ^= 1;
  }
  __syncthreads();

  const int col = nn + wn * 16 + idx;
#pragma unroll
  for (int mt = 0; mt < 2; ++mt) {
    const f32x8& ac = mt ? acc1 : acc0;
#pragma unroll
    for (int r = 0; r < 8; ++r) {
      int m = mt * 16 + r + half * 8;
      int slot = m_base + m;
      if (slot < cnt) {
        int i = Itk[m];                            // i = t*K + k, unique
        yws[(size_t)i * kH + col] = ac[r];         // plain stores, no atomics
      }
    }
  }
}

// ---------------------------------------------------------------------------
// 5) Deterministic reduction over the K=2 slots per token.
// ---------------------------------------------------------------------------
__global__ void reduce_kernel(const float* __restrict__ yws,
                              float* __restrict__ out) {
  int idx = blockIdx.x * blockDim.x + threadIdx.x;
  if (idx >= kT * kH) return;
  int t = idx / kH;
  int h = idx - t * kH;
  float s = 0.f;
#pragma unroll
  for (int k = 0; k < kK; ++k) s += yws[(size_t)(t * kK + k) * kH + h];
  out[idx] = s;
}

// ---------------------------------------------------------------------------
extern "C" void kernel_launch(void* const* d_in, const int* in_sizes, int n_in,
                              void* d_out, int out_size, void* d_ws, size_t ws_size,
                              hipStream_t stream) {
  (void)in_sizes; (void)n_in; (void)out_size; (void)ws_size;
  const float*     x   = (const float*)d_in[0];
  const long long* ids = (const long long*)d_in[1];
  const float*     rw  = (const float*)d_in[2];
  const float*     w1  = (const float*)d_in[3];
  const float*     w2  = (const float*)d_in[4];
  float*           out = (float*)d_out;

  char* ws = (char*)d_ws;
  // Workspace layout (all offsets 32B-aligned):
  size_t off = 0;
  int*    counts = (int*)(ws + off);    off += 256;
  int*    ilist  = (int*)(ws + off);    off += (size_t)kE * MAXT * 4;
  float*  wlist  = (float*)(ws + off);  off += (size_t)kE * MAXT * 4;
  __bf16* px     = (__bf16*)(ws + off); off += (size_t)kT * kH * 2;
  __bf16* pw1    = (__bf16*)(ws + off); off += (size_t)kE * NT1 * KBC1 * 32 * 32;
  __bf16* pw2    = (__bf16*)(ws + off); off += (size_t)kE * NT2 * KBC2 * 32 * 32;
  __bf16* act    = (__bf16*)(ws + off); off += (size_t)kE * MAXT * kI * 2;
  float*  yws    = (float*)(ws + off);

  route_kernel<<<1, 32, 0, stream>>>(ids, rw, counts, ilist, wlist);

  pack_x_kernel<<<(kT * kH / 4 + 255) / 256, 256, 0, stream>>>(x, px);

  {
    int tot1 = kE * NT1 * KBC1 * 32;
    pack_w_kernel<<<(tot1 + 255) / 256, 256, 0, stream>>>(w1, pw1, kN1, NT1, KBC1);
    int tot2 = kE * NT2 * KBC2 * 32;
    pack_w_kernel<<<(tot2 + 255) / 256, 256, 0, stream>>>(w2, pw2, kH, NT2, KBC2);
  }

  gemm1_silu_kernel<<<dim3(kI / 128, MAXT / 32, kE), 256, 0, stream>>>(
      px, (const bf16x16*)pw1, counts, ilist, wlist, act);

  gemm2_scatter_kernel<<<dim3(kH / 128, MAXT / 32, kE), 256, 0, stream>>>(
      act, (const bf16x16*)pw2, counts, ilist, yws);

  reduce_kernel<<<(kT * kH + 255) / 256, 256, 0, stream>>>(yws, out);
}
